// Block_70033736728865
// MI455X (gfx1250) — matrix operand order repaired and checked by reading, not compiled, not run on hardware
//
#include <hip/hip_runtime.h>

// ---------------------------------------------------------------------------
// Problem constants (match reference)
// ---------------------------------------------------------------------------
#define T_TOK   2048
#define D_MODEL 1024
#define N_HEADS 16
#define N_KV    4
#define D_HEAD  64
#define N_EXP   32
#define TOPK_E  4
#define D_FF    1024
#define WIN     128
#define N_QKV   (N_HEADS * D_HEAD + 2 * N_KV * D_HEAD)   // 1536

typedef __attribute__((ext_vector_type(16))) _Float16 v16h;
typedef __attribute__((ext_vector_type(8)))  _Float16 v8h;
typedef __attribute__((ext_vector_type(8)))  float    v8f;
typedef int v4i __attribute__((vector_size(16)));        // 128-bit chunk type

// --- gfx1250 async global->LDS path (guarded; falls back to plain copies) ---
#if defined(__has_builtin)
#  if __has_builtin(__builtin_amdgcn_global_load_async_to_lds_b128)
#    define HAVE_ASYNC_LDS 1
#  endif
#  if __has_builtin(__builtin_amdgcn_s_wait_asynccnt)
#    define HAVE_WAIT_ASYNC 1
#  endif
#endif
#ifndef HAVE_ASYNC_LDS
#  define HAVE_ASYNC_LDS 0
#endif
#ifndef HAVE_WAIT_ASYNC
#  define HAVE_WAIT_ASYNC 0
#endif

#if HAVE_ASYNC_LDS
// Builtin signature (from clang diagnostic): (v4i AS1*, v4i AS3*, imm offset, imm cpol)
#define ASYNC_CP_B128(gsrc, ldst)                                              \
    __builtin_amdgcn_global_load_async_to_lds_b128(                            \
        (__attribute__((address_space(1))) v4i*)(gsrc),                        \
        (__attribute__((address_space(3))) v4i*)(ldst), 0, 0)
#endif

// ---------------------------------------------------------------------------
// f32 [K][N]  ->  f16 [N][K]  (transpose + downconvert; weights for WMMA B)
// ---------------------------------------------------------------------------
__global__ void k_transpose_f16(const float* __restrict__ W, _Float16* __restrict__ Wt,
                                int K, int N) {
    long idx = (long)blockIdx.x * blockDim.x + threadIdx.x;
    long total = (long)K * N;
    if (idx >= total) return;
    int k = (int)(idx / N);
    int n = (int)(idx % N);
    Wt[(size_t)n * K + k] = (_Float16)W[idx];
}

// ---------------------------------------------------------------------------
// RMSNorm: per-token, writes optional f32 copy + f16 copy (GEMM A operand)
// ---------------------------------------------------------------------------
__global__ void k_rmsnorm(const float* __restrict__ x, const float* __restrict__ w,
                          float* __restrict__ o32, _Float16* __restrict__ o16) {
    __shared__ float red[256];
    int t = blockIdx.x, tid = threadIdx.x;
    const float* xr = x + (size_t)t * D_MODEL;
    float s = 0.f;
    for (int d = tid; d < D_MODEL; d += 256) { float v = xr[d]; s += v * v; }
    red[tid] = s; __syncthreads();
    for (int off = 128; off > 0; off >>= 1) {
        if (tid < off) red[tid] += red[tid + off];
        __syncthreads();
    }
    float rms = rsqrtf(red[0] / (float)D_MODEL + 1e-5f);
    for (int d = tid; d < D_MODEL; d += 256) {
        float v = w[d] * xr[d] * rms;
        if (o32) o32[(size_t)t * D_MODEL + d] = v;
        o16[(size_t)t * D_MODEL + d] = (_Float16)v;
    }
}

// ---------------------------------------------------------------------------
// WMMA GEMM:  C[M,N] = A[M,K](f16,row-major) x Bt[N,K](f16, B pre-transposed)
//   mode 0: C = acc (+bias)
//   mode 1: C = acc + bias + resid
//   mode 2: C += rowscale[row] * acc   (MoE combine-weighted accumulation)
//
// Block tile 64x64, 128 threads = 4 waves; wave w owns rows [w*16, w*16+16)
// and all 64 columns (4 accumulators, A fragment reused 4x per K-step).
// A/B 64x32 f16 tiles staged in LDS via GLOBAL_LOAD_ASYNC_TO_LDS_B128
// (ASYNCcnt) when the builtin exists; plain vector copies otherwise.
// Fragment layouts follow CDNA5 ISA 7.12.2 for V_WMMA_F32_16X16X32_F16.
// ---------------------------------------------------------------------------
#define BM 64
#define BN 64
#define BK 32

__global__ void k_gemm_wmma(const _Float16* __restrict__ A, const _Float16* __restrict__ Bt,
                            const float* __restrict__ bias, const float* __restrict__ resid,
                            const float* __restrict__ rowscale, float* __restrict__ C,
                            int M, int N, int K, int mode) {
    __shared__ _Float16 As[BM * BK];   // [64 rows][32 k] row-major
    __shared__ _Float16 Bs[BN * BK];

    int tid  = threadIdx.x;
    int wave = tid >> 5;
    int lane = tid & 31;
    int l15  = lane & 15;   // A row-in-frag / B col-in-frag / D col
    int hi   = lane >> 4;   // K-chunk selector / D row-half selector

    int m0b = blockIdx.y * BM;
    int n0b = blockIdx.x * BN;

    // Staging: 256 16B chunks per tile; thread handles chunks tid and tid+128.
    int r0 = tid >> 2,          p0 = tid & 3;        // rows 0..31
    int r1 = (tid + 128) >> 2,  p1 = tid & 3;        // rows 32..63
    const _Float16* gA0 = A  + (size_t)(m0b + r0) * K + p0 * 8;
    const _Float16* gA1 = A  + (size_t)(m0b + r1) * K + p1 * 8;
    const _Float16* gB0 = Bt + (size_t)(n0b + r0) * K + p0 * 8;
    const _Float16* gB1 = Bt + (size_t)(n0b + r1) * K + p1 * 8;
    _Float16* lA0 = &As[r0 * BK + p0 * 8];
    _Float16* lA1 = &As[r1 * BK + p1 * 8];
    _Float16* lB0 = &Bs[r0 * BK + p0 * 8];
    _Float16* lB1 = &Bs[r1 * BK + p1 * 8];

    v8f acc[4];
#pragma unroll
    for (int j = 0; j < 4; ++j) acc[j] = (v8f){0.f,0.f,0.f,0.f,0.f,0.f,0.f,0.f};

    for (int k0 = 0; k0 < K; k0 += BK) {
#if HAVE_ASYNC_LDS
        ASYNC_CP_B128(gA0 + k0, lA0);
        ASYNC_CP_B128(gA1 + k0, lA1);
        ASYNC_CP_B128(gB0 + k0, lB0);
        ASYNC_CP_B128(gB1 + k0, lB1);
#  if HAVE_WAIT_ASYNC
        __builtin_amdgcn_s_wait_asynccnt(0);
#  else
        asm volatile("s_wait_asynccnt 0x0" ::: "memory");
#  endif
#else
        *(v8h*)lA0 = *(const v8h*)(gA0 + k0);
        *(v8h*)lA1 = *(const v8h*)(gA1 + k0);
        *(v8h*)lB0 = *(const v8h*)(gB0 + k0);
        *(v8h*)lB1 = *(const v8h*)(gB1 + k0);
#endif
        __syncthreads();

        // A fragment for this wave's 16 rows (reused across 4 N-subtiles)
        v16h a;
        const _Float16* par = &As[(wave * 16 + l15) * BK + hi * 8];
#pragma unroll
        for (int i = 0; i < 8; ++i) { a[i] = par[i]; a[i + 8] = par[i + 16]; }

#pragma unroll
        for (int j = 0; j < 4; ++j) {
            v16h b;
            const _Float16* pbr = &Bs[(j * 16 + l15) * BK + hi * 8];
#pragma unroll
            for (int i = 0; i < 8; ++i) { b[i] = pbr[i]; b[i + 8] = pbr[i + 16]; }
            acc[j] = __builtin_amdgcn_wmma_f32_16x16x32_f16(
                         false, a, false, b, (short)0, acc[j], false, false);
        }
        __syncthreads();
    }

#pragma unroll
    for (int j = 0; j < 4; ++j) {
        int col = n0b + j * 16 + l15;
        if (mode == 2) {
#pragma unroll
            for (int r = 0; r < 8; ++r) {
                int row = m0b + wave * 16 + hi * 8 + r;
                size_t o = (size_t)row * N + col;
                C[o] += rowscale[row] * acc[j][r];
            }
        } else {
            float bv = bias ? bias[col] : 0.f;
#pragma unroll
            for (int r = 0; r < 8; ++r) {
                int row = m0b + wave * 16 + hi * 8 + r;
                size_t o = (size_t)row * N + col;
                float v = acc[j][r] + bv;
                if (mode == 1) v += resid[o];
                C[o] = v;
            }
        }
    }
}

// ---------------------------------------------------------------------------
// RoPE: one wave per (token, head); lane = frequency pair p (0..31)
// ---------------------------------------------------------------------------
__global__ void k_rope(const float* __restrict__ qkv,
                       float* __restrict__ qr, float* __restrict__ kr) {
    int t = blockIdx.x, hh = blockIdx.y, p = threadIdx.x;
    float pos  = (float)t / 32.0f;                              // ROPE_SCALE
    float invf = powf(150000.0f, -((float)(2 * p) / 64.0f));    // ROPE_BASE
    float f = pos * invf;
    float sn = sinf(f), cs = cosf(f);
    const float* row = qkv + (size_t)t * N_QKV;
    if (hh < N_HEADS) {
        float x1 = row[hh * D_HEAD + 2 * p];
        float x2 = row[hh * D_HEAD + 2 * p + 1];
        float* o = qr + ((size_t)t * N_HEADS + hh) * D_HEAD;
        o[2 * p]     = x1 * cs - x2 * sn;
        o[2 * p + 1] = x1 * sn + x2 * cs;
    } else {
        int kv = hh - N_HEADS;
        float x1 = row[N_HEADS * D_HEAD + kv * D_HEAD + 2 * p];
        float x2 = row[N_HEADS * D_HEAD + kv * D_HEAD + 2 * p + 1];
        float* o = kr + ((size_t)t * N_KV + kv) * D_HEAD;
        o[2 * p]     = x1 * cs - x2 * sn;
        o[2 * p + 1] = x1 * sn + x2 * cs;
    }
}

// ---------------------------------------------------------------------------
// Sliding-window attention with sink. One wave per (query t, head h);
// lane owns 2 of the 64 head dims; shuffle-reduced dot per key.
// ---------------------------------------------------------------------------
__global__ void k_attn(const float* __restrict__ qr, const float* __restrict__ kr,
                       const float* __restrict__ qkv, const float* __restrict__ sink,
                       _Float16* __restrict__ y16) {
    __shared__ float logit[WIN + 1];
    int t = blockIdx.x, h = blockIdx.y;
    int lane = threadIdx.x;
    int kv = h / (N_HEADS / N_KV);
    int d0 = 2 * lane;

    const float* q = qr + ((size_t)t * N_HEADS + h) * D_HEAD;
    float q0 = q[d0], q1 = q[d0 + 1];

    int jlo = t - WIN; if (jlo < 0) jlo = 0;
    float sv = sink[h];
    float mval = sv;

    for (int j = jlo; j <= t; ++j) {
        const float* kp = kr + ((size_t)j * N_KV + kv) * D_HEAD;
        float p = q0 * kp[d0] + q1 * kp[d0 + 1];
        for (int off = 16; off > 0; off >>= 1) p += __shfl_xor(p, off, 32);
        p *= 0.125f;                           // 1/sqrt(64)
        if (lane == 0) logit[j - jlo] = p;
        mval = fmaxf(mval, p);
    }
    __syncthreads();

    float denom = expf(sv - mval);             // sink contribution
    float a0 = 0.f, a1 = 0.f;
    for (int j = jlo; j <= t; ++j) {
        float w = expf(logit[j - jlo] - mval);
        denom += w;
        const float* vp = qkv + (size_t)j * N_QKV + (N_HEADS * D_HEAD + N_KV * D_HEAD)
                          + kv * D_HEAD;
        a0 += w * vp[d0];
        a1 += w * vp[d0 + 1];
    }
    float inv = 1.f / denom;
    _Float16* o = y16 + (size_t)t * (N_HEADS * D_HEAD) + h * D_HEAD;
    o[d0]     = (_Float16)(a0 * inv);
    o[d0 + 1] = (_Float16)(a1 * inv);
}

// ---------------------------------------------------------------------------
// Router: per-token gate logits (lane = expert), top-4 + softmax on lane 0,
// writes transposed combine weights combT[E][T] (zeros for unselected).
// ---------------------------------------------------------------------------
__global__ void k_gate_topk(const float* __restrict__ h2, const float* __restrict__ gw,
                            const float* __restrict__ gb, float* __restrict__ combT) {
    __shared__ float gl[N_EXP];
    __shared__ float wsel[N_EXP];
    int t = blockIdx.x, e = threadIdx.x;
    const float* hr = h2 + (size_t)t * D_MODEL;
    float acc = gb[e];
    for (int k = 0; k < D_MODEL; ++k)
        acc += hr[k] * gw[(size_t)k * N_EXP + e];   // coalesced across lanes
    gl[e] = acc;
    wsel[e] = 0.f;
    __syncthreads();
    if (e == 0) {
        int   idx[TOPK_E];
        float val[TOPK_E];
        bool  used[N_EXP];
        for (int i = 0; i < N_EXP; ++i) used[i] = false;
        for (int r = 0; r < TOPK_E; ++r) {          // strict '>' matches top_k ties
            float best = -3.4e38f; int bi = 0;
            for (int j = 0; j < N_EXP; ++j)
                if (!used[j] && gl[j] > best) { best = gl[j]; bi = j; }
            used[bi] = true; idx[r] = bi; val[r] = best;
        }
        float mx = val[0];
        float s = 0.f;
        for (int r = 0; r < TOPK_E; ++r) s += expf(val[r] - mx);
        for (int r = 0; r < TOPK_E; ++r) wsel[idx[r]] = expf(val[r] - mx) / s;
    }
    __syncthreads();
    combT[(size_t)e * T_TOK + t] = wsel[e];
}

// ---------------------------------------------------------------------------
// SwiGLU activation: act16 = f16( silu(g) * u ),  ug = [u | g] per token
// ---------------------------------------------------------------------------
__global__ void k_silu_mul(const float* __restrict__ ug, _Float16* __restrict__ a16) {
    long idx = (long)blockIdx.x * blockDim.x + threadIdx.x;
    if (idx >= (long)T_TOK * D_FF) return;
    int n = (int)(idx / D_FF);
    int f = (int)(idx % D_FF);
    float u = ug[(size_t)n * (2 * D_FF) + f];
    float g = ug[(size_t)n * (2 * D_FF) + D_FF + f];
    float s = g / (1.f + expf(-g));
    a16[idx] = (_Float16)(s * u);
}

__global__ void k_copy(const float* __restrict__ src, float* __restrict__ dst, long n) {
    long idx = (long)blockIdx.x * blockDim.x + threadIdx.x;
    if (idx < n) dst[idx] = src[idx];
}

// ---------------------------------------------------------------------------
// Host-side orchestration
// ---------------------------------------------------------------------------
extern "C" void kernel_launch(void* const* d_in, const int* in_sizes, int n_in,
                              void* d_out, int out_size, void* d_ws, size_t ws_size,
                              hipStream_t stream) {
    const float* x      = (const float*)d_in[0];
    const float* anw    = (const float*)d_in[1];
    const float* fnw    = (const float*)d_in[2];
    const float* qkv_w  = (const float*)d_in[3];
    const float* qkv_b  = (const float*)d_in[4];
    const float* proj_w = (const float*)d_in[5];
    const float* proj_b = (const float*)d_in[6];
    const float* sink   = (const float*)d_in[7];
    const float* gate_w = (const float*)d_in[8];
    const float* gate_b = (const float*)d_in[9];
    const float* W_up   = (const float*)d_in[10];
    const float* W_dn   = (const float*)d_in[11];
    float* out = (float*)d_out;

    // Workspace carve (~85 MB total, 256B aligned slices)
    char* p = (char*)d_ws;
    auto carve = [&](size_t bytes) -> char* {
        char* r = p;
        p += (bytes + 255) & ~(size_t)255;
        return r;
    };
    _Float16* h1_16  = (_Float16*)carve((size_t)T_TOK * D_MODEL * 2);
    _Float16* wq_t   = (_Float16*)carve((size_t)N_QKV * D_MODEL * 2);
    float*    qkvb   = (float*)   carve((size_t)T_TOK * N_QKV * 4);
    float*    qrope  = (float*)   carve((size_t)T_TOK * N_HEADS * D_HEAD * 4);
    float*    krope  = (float*)   carve((size_t)T_TOK * N_KV * D_HEAD * 4);
    _Float16* y16    = (_Float16*)carve((size_t)T_TOK * N_HEADS * D_HEAD * 2);
    _Float16* wp_t   = (_Float16*)carve((size_t)D_MODEL * D_MODEL * 2);
    float*    x1     = (float*)   carve((size_t)T_TOK * D_MODEL * 4);
    float*    h2_32  = (float*)   carve((size_t)T_TOK * D_MODEL * 4);
    _Float16* h2_16  = (_Float16*)carve((size_t)T_TOK * D_MODEL * 2);
    float*    combT  = (float*)   carve((size_t)N_EXP * T_TOK * 4);
    _Float16* wup_t  = (_Float16*)carve((size_t)(2 * D_FF) * D_MODEL * 2);
    _Float16* wdn_t  = (_Float16*)carve((size_t)D_MODEL * D_FF * 2);
    float*    ug     = (float*)   carve((size_t)T_TOK * (2 * D_FF) * 4);
    _Float16* act16  = (_Float16*)carve((size_t)T_TOK * D_FF * 2);

    const int tb = 256;

    // ---- Attention half ----
    {   long tot = (long)D_MODEL * N_QKV;
        k_transpose_f16<<<(unsigned)((tot + tb - 1) / tb), tb, 0, stream>>>(qkv_w, wq_t, D_MODEL, N_QKV); }
    k_rmsnorm<<<T_TOK, 256, 0, stream>>>(x, anw, nullptr, h1_16);
    k_gemm_wmma<<<dim3(N_QKV / BN, T_TOK / BM), 128, 0, stream>>>(
        h1_16, wq_t, qkv_b, nullptr, nullptr, qkvb, T_TOK, N_QKV, D_MODEL, 0);
    k_rope<<<dim3(T_TOK, N_HEADS + N_KV), 32, 0, stream>>>(qkvb, qrope, krope);
    k_attn<<<dim3(T_TOK, N_HEADS), 32, 0, stream>>>(qrope, krope, qkvb, sink, y16);
    {   long tot = (long)D_MODEL * D_MODEL;
        k_transpose_f16<<<(unsigned)((tot + tb - 1) / tb), tb, 0, stream>>>(proj_w, wp_t, D_MODEL, D_MODEL); }
    k_gemm_wmma<<<dim3(D_MODEL / BN, T_TOK / BM), 128, 0, stream>>>(
        y16, wp_t, proj_b, x, nullptr, x1, T_TOK, D_MODEL, D_MODEL, 1);

    // d_out starts as the post-attention residual; experts accumulate into it.
    k_copy<<<(unsigned)(((long)T_TOK * D_MODEL + tb - 1) / tb), tb, 0, stream>>>(
        x1, out, (long)T_TOK * D_MODEL);

    // ---- MoE half ----
    k_rmsnorm<<<T_TOK, 256, 0, stream>>>(x1, fnw, h2_32, h2_16);
    k_gate_topk<<<T_TOK, 32, 0, stream>>>(h2_32, gate_w, gate_b, combT);

    for (int e = 0; e < N_EXP; ++e) {
        const float* wu = W_up + (size_t)e * D_MODEL * (2 * D_FF);
        const float* wd = W_dn + (size_t)e * D_FF * D_MODEL;
        {   long tot = (long)D_MODEL * (2 * D_FF);
            k_transpose_f16<<<(unsigned)((tot + tb - 1) / tb), tb, 0, stream>>>(wu, wup_t, D_MODEL, 2 * D_FF); }
        {   long tot = (long)D_FF * D_MODEL;
            k_transpose_f16<<<(unsigned)((tot + tb - 1) / tb), tb, 0, stream>>>(wd, wdn_t, D_FF, D_MODEL); }
        k_gemm_wmma<<<dim3((2 * D_FF) / BN, T_TOK / BM), 128, 0, stream>>>(
            h2_16, wup_t, nullptr, nullptr, nullptr, ug, T_TOK, 2 * D_FF, D_MODEL, 0);
        k_silu_mul<<<(unsigned)(((long)T_TOK * D_FF + tb - 1) / tb), tb, 0, stream>>>(ug, act16);
        k_gemm_wmma<<<dim3(D_MODEL / BN, T_TOK / BM), 128, 0, stream>>>(
            act16, wdn_t, nullptr, nullptr, combT + (size_t)e * T_TOK, out,
            T_TOK, D_MODEL, D_FF, 2);
    }
}
